// ROIPatchExtractor_91216515432665
// MI455X (gfx1250) — matrix operand order repaired and checked by reading, not compiled, not run on hardware
//
#include <hip/hip_runtime.h>
#include <hip/hip_bf16.h>

// ROI patch extractor for MI455X (gfx1250, wave32, WMMA).
//
// out[n,c] = Wy[n] * P[n,c] * Wx[n]^T  (composed bilinear-of-bilinear)
//   - vertical pass: 4-tap gather (sparse Wy) -> f16 strip in LDS (b64 loads)
//   - horizontal pass: dense f16 WMMA GEMM (v_wmma_f32_16x16x32_f16)
// Wx is pre-materialized in WMMA B-fragment lane order in d_ws and loaded
// ONCE per N-tile (shared across the 3 channels). Register budget kept
// < 256 VGPRs and occupancy pinned at >= 4 waves/SIMD for the gather phase.

typedef __attribute__((ext_vector_type(16))) _Float16 v16h;
typedef __attribute__((ext_vector_type(8)))  _Float16 v8h;
typedef __attribute__((ext_vector_type(2)))  _Float16 v2h;
typedef __attribute__((ext_vector_type(8)))  float    v8f;

#define NB   128           // B * 4 patches
#define NC   3
#define HW   448
#define PH   224           // patch size
#define NKT  7             // K tiles: 224 / 32
#define NNT  28            // N tiles: 448 / 16
#define TRPAD 232          // padded LDS row stride (halves), conflict-free
#define IMG_ELEMS ((size_t)NB * NC * HW * HW)

// ---------------------------------------------------------------------------
// Kernel 1a: per-sample, per-output-coordinate composed 4-tap tables.
// For axis y: indices are GLOBAL rows in x (quadrant offset folded in).
// ---------------------------------------------------------------------------
__device__ __forceinline__ void map_u_to_patch(int R, int& i0, int& i1, float& v) {
    // upsampled-patch coord R (0..447) -> patch coords (align_corners=False 2x)
    float su = fminf(fmaxf((float)R * 0.5f - 0.25f, 0.0f), 223.0f);
    i0 = (int)su;
    v  = su - (float)i0;
    i1 = min(i0 + 1, PH - 1);
}

__global__ void taps_kernel(const int* __restrict__ tops,
                            const int* __restrict__ lefts,
                            const int* __restrict__ crop_hs,
                            const int* __restrict__ crop_ws,
                            int* __restrict__ yIdx, float* __restrict__ yW,
                            int* __restrict__ xIdx, float* __restrict__ xW) {
    int t = blockIdx.x * blockDim.x + threadIdx.x;
    const int NTOT = 2 * NB * HW;
    if (t >= NTOT) return;
    int axis = t / (NB * HW);
    int r    = t % (NB * HW);
    int n = r / HW, o = r % HW;
    int q = n & 3;

    int   crop = axis ? crop_ws[n] : crop_hs[n];
    int   off  = axis ? lefts[n]   : tops[n];
    int   pb   = axis ? (q & 1) * PH : (q >> 1) * PH;   // patch offset in x

    float cf = (float)crop;
    float s  = (o + 0.5f) * cf * (1.0f / 448.0f) - 0.5f;
    s = fminf(fmaxf(s, 0.0f), cf - 1.0f);
    int   r0 = (int)s;
    float wr = s - (float)r0;
    int   r1 = min(r0 + 1, crop - 1);
    int   R0 = off + r0, R1 = off + r1;                 // rows in upsampled patch

    int a0, a1, b0, b1; float v0, v1;
    map_u_to_patch(R0, a0, a1, v0);
    map_u_to_patch(R1, b0, b1, v1);

    int*   I = (axis ? xIdx : yIdx) + (size_t)(n * HW + o) * 4;
    float* W = (axis ? xW   : yW)   + (size_t)(n * HW + o) * 4;
    I[0] = pb + a0; I[1] = pb + a1; I[2] = pb + b0; I[3] = pb + b1;
    W[0] = (1.0f - wr) * (1.0f - v0);
    W[1] = (1.0f - wr) * v0;
    W[2] = wr * (1.0f - v1);
    W[3] = wr * v1;
}

// ---------------------------------------------------------------------------
// Kernel 1b: materialize Wx^T in WMMA B-fragment lane layout.
// B (32x16 f16): n = lane%16, k = 16*(lane>=16) + e (contiguous order chosen here, same as A)   (e = 0..15 halves/lane)
// frag[((n*7+kt)*28+nt)*512 + lane*16 + e]
// ---------------------------------------------------------------------------
__global__ void wxfrag_kernel(const int* __restrict__ xIdx,
                              const float* __restrict__ xW,
                              _Float16* __restrict__ frag) {
    int t = blockIdx.x * blockDim.x + threadIdx.x;      // NB*NKT*NNT*32
    if (t >= NB * NKT * NNT * 32) return;
    int lane = t & 31;
    int g    = t >> 5;
    int nt   = g % NNT;
    int kt   = (g / NNT) % NKT;
    int n    = g / (NNT * NKT);

    int qcol = nt * 16 + (lane & 15);                   // output column
    const int*   I = xIdx + (size_t)(n * HW + qcol) * 4;
    const float* W = xW   + (size_t)(n * HW + qcol) * 4;
    int   i0 = I[0], i1 = I[1], i2 = I[2], i3 = I[3];
    float w0 = W[0], w1 = W[1], w2 = W[2], w3 = W[3];

    int abase = (n & 1) * PH + kt * 32 + ((lane >= 16) ? 16 : 0); // global col

    v16h h;
#pragma unroll
    for (int e = 0; e < 16; ++e) {
        int a = abase + e;
        float w = 0.0f;
        if (i0 == a) w += w0;
        if (i1 == a) w += w1;
        if (i2 == a) w += w2;
        if (i3 == a) w += w3;
        h[e] = (_Float16)w;
    }
    *(v16h*)(frag + ((size_t)((n * NKT + kt) * NNT + nt) * 512 + lane * 16)) = h;
}

// ---------------------------------------------------------------------------
// Kernel 2: fused strip kernel. One block = (sample n, 16 output rows).
// Phase 1: vertical 4-tap gather (float2 loads) -> f16 T[3][16][224] in LDS.
// Phase 2: out_strip = T * WxFrag via v_wmma_f32_16x16x32_f16.
//          N-tile outer / channel inner: B-fragments loaded once per tile.
// ---------------------------------------------------------------------------
__global__ void __launch_bounds__(128)
__attribute__((amdgpu_waves_per_eu(4)))
roi_wmma_kernel(const float* __restrict__ x,
                const int* __restrict__ yIdx, const float* __restrict__ yW,
                const _Float16* __restrict__ frag, float* __restrict__ out) {
    __shared__ _Float16 Tr[NC * 16 * TRPAD];            // 22,272 B

    int blk = blockIdx.x;
    int n   = blk / NNT;
    int o0  = (blk % NNT) * 16;
    int b   = n >> 2, q = n & 3;
    int qc  = (q & 1) * PH;                             // global col offset of patch
    int tid = threadIdx.x;

    // ---- Phase 1: vertical pass, two adjacent columns per thread ----
    const int HALFW = PH / 2;                           // 112 column-pairs
    for (int t = tid; t < NC * 16 * HALFW; t += 128) {
        int c  = t / (16 * HALFW);
        int rm = t % (16 * HALFW);
        int mm = rm / HALFW, a = (rm % HALFW) * 2;
        int o  = o0 + mm;
        const int*   I = yIdx + (size_t)(n * HW + o) * 4;
        const float* W = yW   + (size_t)(n * HW + o) * 4;
        const float* xc = x + ((size_t)(b * NC + c) * HW) * HW + (qc + a);
        float2 p0 = *(const float2*)(xc + (size_t)I[0] * HW);
        float2 p1 = *(const float2*)(xc + (size_t)I[1] * HW);
        float2 p2 = *(const float2*)(xc + (size_t)I[2] * HW);
        float2 p3 = *(const float2*)(xc + (size_t)I[3] * HW);
        float va = W[0] * p0.x + W[1] * p1.x + W[2] * p2.x + W[3] * p3.x;
        float vb = W[0] * p0.y + W[1] * p1.y + W[2] * p2.y + W[3] * p3.y;
        v2h pk; pk[0] = (_Float16)va; pk[1] = (_Float16)vb;
        *(v2h*)(&Tr[(c * 16 + mm) * TRPAD + a]) = pk;   // 4B-aligned b32 store
    }
    __syncthreads();

    // ---- Phase 2: horizontal pass as WMMA GEMM ----
    int wave = tid >> 5, lane = tid & 31;
    int m     = lane & 15;
    int koffA = (lane >> 4) * 16;    // A k order chosen here (matches B frag): k = e + 16*(lane>=16)

    int col     = (lane & 15);
    int rowbase = o0 + ((lane >> 4) << 3);

#pragma unroll 1
    for (int ntl = 0; ntl < NNT / 4; ++ntl) {
        int nt = wave * (NNT / 4) + ntl;
        const _Float16* bp =
            frag + ((size_t)(n * NKT) * NNT + nt) * 512 + lane * 16;

        // B-fragments for this N-tile: loaded ONCE, reused by all 3 channels.
        v16h Bf[NKT];
#pragma unroll
        for (int kt = 0; kt < NKT; ++kt)
            Bf[kt] = *(const v16h*)(bp + (size_t)kt * NNT * 512);

#pragma unroll 1
        for (int c = 0; c < NC; ++c) {
            const _Float16* trc = Tr + (c * 16 + m) * TRPAD;

            v16h A[NKT];
#pragma unroll
            for (int kt = 0; kt < NKT; ++kt) {
                const _Float16* ap = trc + kt * 32 + koffA;
                v8h lo = *(const v8h*)(ap);        // k = base..base+7
                v8h hi = *(const v8h*)(ap + 8);    // k = base+8..base+15
#pragma unroll
                for (int e = 0; e < 8; ++e) { A[kt][e] = lo[e]; A[kt][e + 8] = hi[e]; }
            }

            v8f acc = {};
#pragma unroll
            for (int kt = 0; kt < NKT; ++kt)
                acc = __builtin_amdgcn_wmma_f32_16x16x32_f16(
                    false, A[kt], false, Bf[kt], (short)0, acc, false, false);

            // D layout: M = e + 8*(lane>=16), N = lane%16 -> coalesced row stores
            float* op = out + ((size_t)(n * NC + c) * HW + rowbase) * HW
                            + nt * 16 + col;
#pragma unroll
            for (int e = 0; e < 8; ++e) op[(size_t)e * HW] = acc[e];
        }
    }
}

// ---------------------------------------------------------------------------
// Kernel 3: patch indices (second tuple output), appended after the image.
// ---------------------------------------------------------------------------
__global__ void idx_kernel(float* __restrict__ out) {
    int i = threadIdx.x;
    if (i < NB) out[IMG_ELEMS + i] = (float)(i & 3);
}

// ---------------------------------------------------------------------------
extern "C" void kernel_launch(void* const* d_in, const int* in_sizes, int n_in,
                              void* d_out, int out_size, void* d_ws, size_t ws_size,
                              hipStream_t stream) {
    const float* x       = (const float*)d_in[0];
    const int*   tops    = (const int*)d_in[1];
    const int*   lefts   = (const int*)d_in[2];
    const int*   crop_hs = (const int*)d_in[3];
    const int*   crop_ws = (const int*)d_in[4];
    float*       out     = (float*)d_out;

    char* wsb = (char*)d_ws;
    const size_t FRAG_BYTES = (size_t)NB * NKT * NNT * 512 * sizeof(_Float16); // 25.7 MB
    const size_t TAP_BYTES  = (size_t)NB * HW * 4 * 4;                         // 0.92 MB each

    _Float16* frag = (_Float16*)wsb;
    int*      yI   = (int*)  (wsb + FRAG_BYTES);
    float*    yWp  = (float*)(wsb + FRAG_BYTES + TAP_BYTES);
    int*      xI   = (int*)  (wsb + FRAG_BYTES + 2 * TAP_BYTES);
    float*    xWp  = (float*)(wsb + FRAG_BYTES + 3 * TAP_BYTES);

    taps_kernel<<<(2 * NB * HW + 255) / 256, 256, 0, stream>>>(
        tops, lefts, crop_hs, crop_ws, yI, yWp, xI, xWp);

    wxfrag_kernel<<<(NB * NKT * NNT * 32 + 127) / 128, 128, 0, stream>>>(
        xI, xWp, frag);

    roi_wmma_kernel<<<NB * NNT, 128, 0, stream>>>(x, yI, yWp, frag, out);

    if ((size_t)out_size >= IMG_ELEMS + NB)
        idx_kernel<<<1, 128, 0, stream>>>(out);
}
